// NeighborhoodCrossAttentionV2_82231443849836
// MI455X (gfx1250) — matrix operand order repaired and checked by reading, not compiled, not run on hardware
//
#include <hip/hip_runtime.h>
#include <math.h>

typedef __attribute__((ext_vector_type(16))) _Float16 v16h;
typedef __attribute__((ext_vector_type(8)))  _Float16 v8h;
typedef __attribute__((ext_vector_type(8)))  float    v8f;

// Problem constants (fixed by reference setup_inputs)
#define BATCH 2
#define NCLS  20
#define DIM   64
#define HEADS 4
#define HDIM  16
#define HH    48
#define WW    48
#define ZZ    8
#define NPOS  (HH * WW * ZZ)          // 18432
#define BN    (BATCH * NPOS)          // 36864
#define EPSV  1e-5f

// ---------------------------------------------------------------------------
// Kernel 1: instance-norm statistics for x (B*20 channels) and skip (B*64).
// ---------------------------------------------------------------------------
__global__ __launch_bounds__(256)
void stats_kernel(const float* __restrict__ x, const float* __restrict__ skip,
                  float* __restrict__ mu_x, float* __restrict__ rs_x,
                  float* __restrict__ mu_s, float* __restrict__ rs_s) {
    const int c   = blockIdx.x;
    const int tid = threadIdx.x;
    const int nx  = BATCH * NCLS;
    const float* src;
    float *mu, *rs;
    if (c < nx) { src = x    + (size_t)c * NPOS;  mu = mu_x + c;  rs = rs_x + c; }
    else        { int cc = c - nx;
                  src = skip + (size_t)cc * NPOS; mu = mu_s + cc; rs = rs_s + cc; }
    float s = 0.f, s2 = 0.f;
    for (int i = tid; i < NPOS; i += 256) { float v = src[i]; s += v; s2 += v * v; }
    __shared__ float red0[256];
    __shared__ float red1[256];
    red0[tid] = s; red1[tid] = s2;
    __syncthreads();
    for (int off = 128; off > 0; off >>= 1) {
        if (tid < off) { red0[tid] += red0[tid + off]; red1[tid] += red1[tid + off]; }
        __syncthreads();
    }
    if (tid == 0) {
        float m   = red0[0] * (1.0f / NPOS);
        float var = red1[0] * (1.0f / NPOS) - m * m;
        *mu = m;
        *rs = rsqrtf(var + EPSV);
    }
}

// ---------------------------------------------------------------------------
// WMMA fragment helpers (CDNA5 wave32 layouts, CDNA5 ISA §7.12.2).
// Both fragments load as 2x 16-byte LDS vectors (contiguous-K storage).
// ---------------------------------------------------------------------------
union frag16 { v16h v; v8h h[2]; };

// A (16x32 MxK): lane(0-15)=row m, halves j<8 -> K=hi*8+j, j>=8 -> K=16+hi*8+(j-8)
__device__ __forceinline__ v16h load_fragA(const _Float16* row, int hi, int kb) {
    frag16 A;
    A.h[0] = *(const v8h*)(row + kb + hi * 8);
    A.h[1] = *(const v8h*)(row + kb + 16 + hi * 8);
    return A.v;
}
// B (32x16 KxN) stored transposed (lds_bt[n][k]): j<8 -> K=hi*8+j, j>=8 -> K=16+hi*8+(j-8)
__device__ __forceinline__ v16h load_fragB(const _Float16* colrow, int hi, int kb) {
    frag16 Bf;
    const _Float16* p = colrow + kb + hi * 8;
    Bf.h[0] = *(const v8h*)(p);
    Bf.h[1] = *(const v8h*)(p + 16);
    return Bf.v;
}

#define LDA_Q 72   // 64 + 8 pad (36-dword stride -> conflict-free)
#define LDB_Q 72
#define LDA_K 40   // 32 + 8 pad
#define LDB_K 40

// ---------------------------------------------------------------------------
// Kernel 2: q = instnorm(skip) @ q_w^T + q_b, scaled 0.25; stored f16.
// 128 threads = 4 waves; wave w -> column tile w*16; two K=32 WMMAs.
// ---------------------------------------------------------------------------
__global__ __launch_bounds__(128)
void qproj_kernel(const float* __restrict__ skip,
                  const float* __restrict__ mu_s, const float* __restrict__ rs_s,
                  const float* __restrict__ q_w,  const float* __restrict__ q_b,
                  _Float16* __restrict__ qf) {
    __shared__ __align__(16) _Float16 lds_a[16 * LDA_Q];
    __shared__ __align__(16) _Float16 lds_bt[64 * LDB_Q];
    const int tid = threadIdx.x;
    const int r0  = blockIdx.x * 16;
    const int b   = r0 / NPOS;
    const int n0  = r0 - b * NPOS;

    for (int t = tid; t < 64 * 64; t += 128) {        // natural layout = B^T
        int j = t >> 6, k = t & 63;
        lds_bt[j * LDB_Q + k] = (_Float16)q_w[t];
    }
    for (int t = tid; t < 16 * 64; t += 128) {        // normalized A tile
        int mi = t >> 6, c = t & 63;
        int ch = b * DIM + c;
        float v = (skip[(size_t)ch * NPOS + n0 + mi] - mu_s[ch]) * rs_s[ch];
        lds_a[mi * LDA_Q + c] = (_Float16)v;
    }
    __syncthreads();

    const int lane  = tid & 31;               // wave32
    const int m     = lane & 15;
    const int hi    = lane >> 4;
    const int jbase = (tid >> 5) * 16;

    const _Float16* arow = lds_a  + m * LDA_Q;
    const _Float16* brow = lds_bt + (jbase + m) * LDB_Q;
    v8f acc = {};
#pragma unroll
    for (int kb = 0; kb < 64; kb += 32)
        acc = __builtin_amdgcn_wmma_f32_16x16x32_f16(false, load_fragA(arow, hi, kb),
                                                     false, load_fragB(brow, hi, kb),
                                                     (short)0, acc, false, false);
    union { v8f v; float e[8]; } C; C.v = acc;
    const int   jcol = jbase + m;
    const float bias = q_b[jcol];
#pragma unroll
    for (int r = 0; r < 8; ++r) {
        int mm = r + hi * 8;
        qf[(size_t)(r0 + mm) * DIM + jcol] = (_Float16)((C.e[r] + bias) * 0.25f);
    }
}

// ---------------------------------------------------------------------------
// Kernel 3: kv = instnorm(x) @ kv_w^T + kv_b -> kf/vf (f16).
// 256 threads = 8 waves; one K=32 WMMA (K=20 zero-padded).
// ---------------------------------------------------------------------------
__global__ __launch_bounds__(256)
void kvproj_kernel(const float* __restrict__ x,
                   const float* __restrict__ mu_x, const float* __restrict__ rs_x,
                   const float* __restrict__ kv_w, const float* __restrict__ kv_b,
                   _Float16* __restrict__ kf, _Float16* __restrict__ vf) {
    __shared__ __align__(16) _Float16 lds_a[16 * LDA_K];
    __shared__ __align__(16) _Float16 lds_bt[128 * LDB_K];
    const int tid = threadIdx.x;
    const int r0  = blockIdx.x * 16;
    const int b   = r0 / NPOS;
    const int n0  = r0 - b * NPOS;

    for (int t = tid; t < 128 * 32; t += 256) {
        int j = t >> 5, k = t & 31;
        lds_bt[j * LDB_K + k] = (k < NCLS) ? (_Float16)kv_w[j * NCLS + k] : (_Float16)0.0f;
    }
    for (int t = tid; t < 16 * 32; t += 256) {
        int mi = t >> 5, c = t & 31;
        _Float16 v = (_Float16)0.0f;
        if (c < NCLS) {
            int ch = b * NCLS + c;
            v = (_Float16)((x[(size_t)ch * NPOS + n0 + mi] - mu_x[ch]) * rs_x[ch]);
        }
        lds_a[mi * LDA_K + c] = v;
    }
    __syncthreads();

    const int lane  = tid & 31;
    const int m     = lane & 15;
    const int hi    = lane >> 4;
    const int jbase = (tid >> 5) * 16;

    v8f acc = {};
    acc = __builtin_amdgcn_wmma_f32_16x16x32_f16(false, load_fragA(lds_a + m * LDA_K, hi, 0),
                                                 false, load_fragB(lds_bt + (jbase + m) * LDB_K, hi, 0),
                                                 (short)0, acc, false, false);
    union { v8f v; float e[8]; } C; C.v = acc;
    const int   jcol = jbase + m;
    const float bias = kv_b[jcol];
#pragma unroll
    for (int r = 0; r < 8; ++r) {
        int      mm  = r + hi * 8;
        _Float16 val = (_Float16)(C.e[r] + bias);
        if (jcol < DIM) kf[(size_t)(r0 + mm) * DIM + jcol]         = val;
        else            vf[(size_t)(r0 + mm) * DIM + (jcol - DIM)] = val;
    }
}

// ---------------------------------------------------------------------------
// Kernel 4: neighborhood attention (f16 q/k/v, f32 math). One thread per
// (b,n,head); prefetch v-rows (global_prefetch_b8) during logit pass.
// ---------------------------------------------------------------------------
__global__ __launch_bounds__(256)
void attn_kernel(const _Float16* __restrict__ qf, const _Float16* __restrict__ kf,
                 const _Float16* __restrict__ vf, const float* __restrict__ rpb,
                 _Float16* __restrict__ ov) {
    const int gid = blockIdx.x * 256 + threadIdx.x;   // ((b*N+n)*HEADS + h)
    if (gid >= BN * HEADS) return;
    const int h  = gid & (HEADS - 1);
    const int bn = gid >> 2;
    const int n  = bn % NPOS;
    const int b  = bn / NPOS;

    const int hp = n / (WW * ZZ);
    const int rm = n - hp * (WW * ZZ);
    const int wp = rm / ZZ;
    const int zp = rm - wp * ZZ;
    const int sh = min(max(hp - 1, 0), HH - 3);
    const int sw = min(max(wp - 1, 0), WW - 3);
    const int sz = min(max(zp - 1, 0), ZZ - 3);

    float qr[HDIM];
    const _Float16* qp = qf + (size_t)bn * DIM + h * HDIM;
#pragma unroll
    for (int d = 0; d < HDIM; ++d) qr[d] = (float)qp[d];

    float lg[27];
    int   np[27];
    float mx = -1e30f;
    int kk = 0;
#pragma unroll
    for (int dh = 0; dh < 3; ++dh)
#pragma unroll
        for (int dw = 0; dw < 3; ++dw)
#pragma unroll
            for (int dz = 0; dz < 3; ++dz, ++kk) {
                const int ih = sh + dh, iw = sw + dw, iz = sz + dz;
                const int nb = (ih * WW + iw) * ZZ + iz;
                const int rel = ((ih - hp + 2) * 5 + (iw - wp + 2)) * 5 + (iz - zp + 2);
                const size_t roff = ((size_t)b * NPOS + nb) * DIM + h * HDIM;
                const _Float16* kp = kf + roff;
                __builtin_prefetch((const void*)(vf + roff), 0, 1);  // global_prefetch_b8
                float s = rpb[h * 125 + rel];
#pragma unroll
                for (int d = 0; d < HDIM; ++d) s += qr[d] * (float)kp[d];
                lg[kk] = s;
                np[kk] = nb;
                mx = fmaxf(mx, s);
            }

    float acc[HDIM];
#pragma unroll
    for (int d = 0; d < HDIM; ++d) acc[d] = 0.f;
    float sum = 0.f;
#pragma unroll
    for (int k2 = 0; k2 < 27; ++k2) {
        const float w = __expf(lg[k2] - mx);
        sum += w;
        const _Float16* vp = vf + ((size_t)b * NPOS + np[k2]) * DIM + h * HDIM;
#pragma unroll
        for (int d = 0; d < HDIM; ++d) acc[d] += w * (float)vp[d];
    }
    const float inv = 1.0f / sum;
    _Float16* op = ov + (size_t)bn * DIM + h * HDIM;
#pragma unroll
    for (int d = 0; d < HDIM; ++d) op[d] = (_Float16)(acc[d] * inv);
}

// ---------------------------------------------------------------------------
// Kernel 5: out = ov @ proj_w^T + proj_b, channel-major f32 output.
// ---------------------------------------------------------------------------
__global__ __launch_bounds__(128)
void oproj_kernel(const _Float16* __restrict__ ov,
                  const float* __restrict__ proj_w, const float* __restrict__ proj_b,
                  float* __restrict__ out) {
    __shared__ __align__(16) _Float16 lds_a[16 * LDA_Q];
    __shared__ __align__(16) _Float16 lds_bt[64 * LDB_Q];
    const int tid = threadIdx.x;
    const int r0  = blockIdx.x * 16;
    const int b   = r0 / NPOS;
    const int n0  = r0 - b * NPOS;

    for (int t = tid; t < 64 * 64; t += 128) {
        int j = t >> 6, k = t & 63;
        lds_bt[j * LDB_Q + k] = (_Float16)proj_w[t];
    }
    for (int t = tid; t < 16 * 64; t += 128) {        // convert-free f16 copy
        int mi = t >> 6, c = t & 63;
        lds_a[mi * LDA_Q + c] = ov[(size_t)(r0 + mi) * DIM + c];
    }
    __syncthreads();

    const int lane  = tid & 31;
    const int m     = lane & 15;
    const int hi    = lane >> 4;
    const int jbase = (tid >> 5) * 16;

    const _Float16* arow = lds_a  + m * LDA_Q;
    const _Float16* brow = lds_bt + (jbase + m) * LDB_Q;
    v8f acc = {};
#pragma unroll
    for (int kb = 0; kb < 64; kb += 32)
        acc = __builtin_amdgcn_wmma_f32_16x16x32_f16(false, load_fragA(arow, hi, kb),
                                                     false, load_fragB(brow, hi, kb),
                                                     (short)0, acc, false, false);
    union { v8f v; float e[8]; } C; C.v = acc;
    const int   jcol = jbase + m;
    const float bias = proj_b[jcol];
#pragma unroll
    for (int r = 0; r < 8; ++r) {
        int mm = r + hi * 8;
        out[((size_t)b * DIM + jcol) * NPOS + (n0 + mm)] = C.e[r] + bias;
    }
}

// ---------------------------------------------------------------------------
extern "C" void kernel_launch(void* const* d_in, const int* in_sizes, int n_in,
                              void* d_out, int out_size, void* d_ws, size_t ws_size,
                              hipStream_t stream) {
    const float* x      = (const float*)d_in[0];
    const float* skip   = (const float*)d_in[1];
    const float* q_w    = (const float*)d_in[2];
    const float* q_b    = (const float*)d_in[3];
    const float* kv_w   = (const float*)d_in[4];
    const float* kv_b   = (const float*)d_in[5];
    const float* proj_w = (const float*)d_in[6];
    const float* proj_b = (const float*)d_in[7];
    const float* rpb    = (const float*)d_in[8];
    float* out = (float*)d_out;

    float* ws   = (float*)d_ws;
    float* mu_x = ws;                      // B*20
    float* rs_x = mu_x + BATCH * NCLS;     // B*20
    float* mu_s = rs_x + BATCH * NCLS;     // B*64
    float* rs_s = mu_s + BATCH * DIM;      // B*64

    _Float16* hbase = (_Float16*)(ws + 512);
    _Float16* qf = hbase;                  // BN*64 halves each
    _Float16* kf = qf + (size_t)BN * DIM;
    _Float16* vf = kf + (size_t)BN * DIM;
    _Float16* ov = vf + (size_t)BN * DIM;

    stats_kernel<<<BATCH * (NCLS + DIM), 256, 0, stream>>>(x, skip, mu_x, rs_x, mu_s, rs_s);
    qproj_kernel<<<BN / 16, 128, 0, stream>>>(skip, mu_s, rs_s, q_w, q_b, qf);
    kvproj_kernel<<<BN / 16, 256, 0, stream>>>(x, mu_x, rs_x, kv_w, kv_b, kf, vf);
    attn_kernel<<<(BN * HEADS) / 256, 256, 0, stream>>>(qf, kf, vf, rpb, ov);
    oproj_kernel<<<BN / 16, 128, 0, stream>>>(ov, proj_w, proj_b, out);
}
